// GCN_70961449665143
// MI455X (gfx1250) — matrix operand order repaired and checked by reading, not compiled, not run on hardware
//
#include <hip/hip_runtime.h>

// ---------------------------------------------------------------------------
// 2-layer GCN forward for MI455X (gfx1250, wave32, WMMA 16x16x32 bf16)
//   support1 = x @ W1                  (LDS-staged WMMA bf16 -> f32)
//   agg1     = segment_sum(w_e * support1[src], dst)  (gather + f32 atomics)
//   h        = relu(agg1 + b1)         (fused into GEMM2 A-operand load)
//   support2 = h @ W2 (10 -> pad 16)   (LDS-staged WMMA bf16 -> f32)
//   out      = b2 + segment_sum(w_e * support2[src], dst)
//
// Workspace rows are padded to the GEMM block tiles so every WMMA store is
// unguarded (no exec-mask ladders); padding rows are never read by SpMM.
// ---------------------------------------------------------------------------

#define GCN_N_NODES   100000
#define GCN_N_EDGES   1600000
#define GCN_N_FEAT    512
#define GCN_N_HID     128
#define GCN_N_CLASSES 10
#define GCN_N_PAD     16       // padded class dim for the 16x16 WMMA tile

#define G1_BM 64               // GEMM1 rows per block
#define G1_BK 32
#define G2_BM 128              // GEMM2 rows per block

#define GCN_ROWS1_PAD 100032   // ceil(N/64)*64   -> S1 rows
#define GCN_ROWS2_PAD 100096   // ceil(N/128)*128 -> AGG / S2 rows

typedef __attribute__((ext_vector_type(16))) __bf16 v16bf;
typedef __attribute__((ext_vector_type(8)))  __bf16 v8bf;
typedef __attribute__((ext_vector_type(8)))  float  v8f;

__device__ __forceinline__ unsigned pack_bf16(float x, float y) {
  union { __bf16 h[2]; unsigned u; } u;
  u.h[0] = (__bf16)x;   // lowers to v_cvt_pk_bf16_f32
  u.h[1] = (__bf16)y;
  return u.u;
}

__device__ __forceinline__ v16bf join8(v8bf lo, v8bf hi) {
  v16bf r;
#pragma unroll
  for (int i = 0; i < 8; ++i) { r[i] = lo[i]; r[8 + i] = hi[i]; }
  return r;
}

// ---------------------------------------------------------------------------
// GEMM1: S1[R1,128] = x[N,512] @ W1[512,128]
//
// Block = 256 threads = 8 waves. Block tile: 64 rows x 128 cols.
// Per k-step (BK=32): cooperatively stage A (64x32, row-major bf16) and
// B (32x128, column-major bf16: each column's K-run contiguous) into LDS,
// then each wave computes a 16-col slab over 4 row-tiles:
// 4 x v_wmma_f32_16x16x32_bf16 per k-step, fragments via ds_load_b128.
//
// Fragment k order used here (contiguous, same for A and B):
//   A lane(row=r, half): elems 0-15 = K[16h..16h+15]
//   B lane(col=r, half): elems 0-15 = K[16h..16h+15]
// Both are contiguous 16B runs in LDS -> two ds_load_b128 per fragment.
// ---------------------------------------------------------------------------
__global__ __launch_bounds__(256) void gcn_gemm1_wmma(
    const float* __restrict__ X,    // [N,512]
    const float* __restrict__ W1,   // [512,128]
    float* __restrict__ S1,         // [R1,128] padded
    int nrows)
{
  __shared__ __align__(16) __bf16 As[G1_BM][G1_BK];      // 4 KB, row-major
  __shared__ __align__(16) __bf16 Bs[GCN_N_HID][G1_BK];  // 8 KB, col-major

  const int tid  = threadIdx.x;
  const int lane = tid & 31;
  const int wave = tid >> 5;            // 0..7 -> 16-col slab
  const int rowBase = blockIdx.x * G1_BM;
  const int colBase = wave * 16;

  const int r    = lane & 15;           // A: row-in-tile, B/C/D: col-in-tile
  const int half = lane >> 4;

  // staging coords
  const int aRow  = tid >> 2;           // 0..63
  const int aKseg = (tid & 3) * 8;      // 8 K values per thread
  const int bCol  = tid & 127;          // 0..127
  const int bKsel = tid >> 7;           // 0..1 -> K half (16 each)

  int aSrcRow = rowBase + aRow;
  if (aSrcRow >= nrows) aSrcRow = nrows - 1;   // clamp: X has only N rows
  const float* __restrict__ xrow = X + (size_t)aSrcRow * GCN_N_FEAT;

  v8f acc[4] = {};

  for (int k0 = 0; k0 < GCN_N_FEAT; k0 += G1_BK) {
    // ---- stage A: 64x32 f32 -> bf16, row-major ----
    unsigned* aDst = (unsigned*)&As[aRow][aKseg];
#pragma unroll
    for (int j = 0; j < 4; ++j) {
      const float2 v = *(const float2*)(xrow + k0 + aKseg + 2 * j);
      aDst[j] = pack_bf16(v.x, v.y);
    }
    // ---- stage B: 32x128 f32 -> bf16, column-major (Bs[col][k]) ----
    unsigned* bDst = (unsigned*)&Bs[bCol][bKsel * 16];
#pragma unroll
    for (int j = 0; j < 8; ++j) {
      const int kk = k0 + bKsel * 16 + 2 * j;
      const float w0 = W1[(size_t)kk * GCN_N_HID + bCol];
      const float w1 = W1[(size_t)(kk + 1) * GCN_N_HID + bCol];
      bDst[j] = pack_bf16(w0, w1);
    }
    __syncthreads();

    // ---- B fragment: Bs[colBase+r][16h .. 16h+15] = 2 x ds_load_b128 ----
    const __bf16* bp = &Bs[colBase + r][16 * half];
    const v16bf b = join8(*(const v8bf*)bp, *(const v8bf*)(bp + 8));

    // ---- 4 row-tiles: A fragment + WMMA each (B reused) ----
#pragma unroll
    for (int rt = 0; rt < 4; ++rt) {
      const __bf16* ap0 = &As[rt * 16 + r][16 * half];
      const __bf16* ap1 = &As[rt * 16 + r][16 * half + 8];
      const v16bf a = join8(*(const v8bf*)ap0, *(const v8bf*)ap1);
      acc[rt] = __builtin_amdgcn_wmma_f32_16x16x32_bf16(
          /*neg_a=*/false, a, /*neg_b=*/false, b,
          /*c_mod=*/(short)0, acc[rt], /*reuse_a=*/false, /*reuse_b=*/false);
    }
    __syncthreads();
  }

  // C/D layout: VGPR t -> M = rt*16 + t + 8*half, N = r. S1 is row-padded
  // to a multiple of 64, so stores need no guard (no exec-mask ladders).
#pragma unroll
  for (int rt = 0; rt < 4; ++rt) {
#pragma unroll
    for (int t = 0; t < 8; ++t) {
      const int m = rowBase + rt * 16 + t + 8 * half;
      S1[(size_t)m * GCN_N_HID + colBase + r] = acc[rt][t];
    }
  }
}

// ---------------------------------------------------------------------------
// SpMM layer 1: AGG[dst] += w_e * S1[src]; one wave per edge, 4 floats/lane.
// ---------------------------------------------------------------------------
__global__ __launch_bounds__(256) void gcn_spmm1(
    const int* __restrict__ src, const int* __restrict__ dst,
    const float* __restrict__ w, const float* __restrict__ S1,
    float* __restrict__ AGG, int nedges)
{
  const int lane = threadIdx.x & 31;
  const int e = blockIdx.x * 8 + (threadIdx.x >> 5);
  if (e >= nedges) return;
  const int   s  = src[e];
  const int   d  = dst[e];
  const float we = w[e];
  const float4 v = *(const float4*)(S1 + (size_t)s * GCN_N_HID + lane * 4);
  float* out = AGG + (size_t)d * GCN_N_HID + lane * 4;
  atomicAdd(out + 0, we * v.x);
  atomicAdd(out + 1, we * v.y);
  atomicAdd(out + 2, we * v.z);
  atomicAdd(out + 3, we * v.w);
}

// ---------------------------------------------------------------------------
// GEMM2: S2[R2,16] = relu(AGG + b1) @ W2pad[128,16]  (cols 10..15 zero).
// Block = 8 waves = 128 rows; wave w owns rows [base+16w, base+16w+16).
// W2pad (bf16, col-major) and b1 (f32) staged in LDS once; AGG is padded
// and zero-filled so loads and stores are both unguarded -> EXEC all-ones.
// K = 128 -> 4 WMMAs per wave; B fragments via ds_load_b128.
// ---------------------------------------------------------------------------
__global__ __launch_bounds__(256) void gcn_gemm2_wmma(
    const float* __restrict__ AGG,  // [R2,128] padded+zeroed
    const float* __restrict__ b1,   // [128]
    const float* __restrict__ W2,   // [128,10]
    float* __restrict__ S2)         // [R2,16] padded
{
  __shared__ __align__(16) __bf16 W2s[GCN_N_PAD][GCN_N_HID]; // 4 KB col-major
  __shared__ __align__(16) float  b1s[GCN_N_HID];            // 512 B

  const int tid  = threadIdx.x;
  const int lane = tid & 31;
  const int wave = tid >> 5;
  const int rowBase = blockIdx.x * G2_BM + wave * 16;

  const int r    = lane & 15;
  const int half = lane >> 4;

  // ---- one-time staging: W2 zero-padded to 16 cols, K-contiguous per col
  {
    const int col  = tid >> 4;        // 0..15
    const int kseg = (tid & 15) * 8;  // 8 K values per thread
    unsigned* dst = (unsigned*)&W2s[col][kseg];
#pragma unroll
    for (int j = 0; j < 4; ++j) {
      const int kk = kseg + 2 * j;
      const float w0 = (col < GCN_N_CLASSES) ? W2[(size_t)kk * GCN_N_CLASSES + col] : 0.0f;
      const float w1 = (col < GCN_N_CLASSES) ? W2[(size_t)(kk + 1) * GCN_N_CLASSES + col] : 0.0f;
      dst[j] = pack_bf16(w0, w1);
    }
    if (tid < GCN_N_HID) b1s[tid] = b1[tid];
  }
  __syncthreads();

  const float* __restrict__ arow = AGG + (size_t)(rowBase + r) * GCN_N_HID;

  v8f acc = {};
  for (int k0 = 0; k0 < GCN_N_HID; k0 += 32) {
    v16bf a, b;
    // A fragment: relu(agg + b1) computed on the fly (bias from LDS)
#pragma unroll
    for (int t = 0; t < 8; ++t) {
      const int kk = k0 + 16 * half + 2 * t;
      const float2 av = *(const float2*)(arow + kk);
      const float2 bv = *(const float2*)(&b1s[kk]);
      a[2 * t]     = (__bf16)fmaxf(av.x + bv.x, 0.0f);
      a[2 * t + 1] = (__bf16)fmaxf(av.y + bv.y, 0.0f);
    }
    // B fragment: W2s[r][k0+16h .. +15] = 2 x ds_load_b128
    const __bf16* bp = &W2s[r][k0 + 16 * half];
    b = join8(*(const v8bf*)bp, *(const v8bf*)(bp + 8));

    acc = __builtin_amdgcn_wmma_f32_16x16x32_bf16(
        false, a, false, b, (short)0, acc, false, false);
  }

  // S2 is row-padded to a multiple of 128 -> unguarded stores.
#pragma unroll
  for (int t = 0; t < 8; ++t) {
    const int m = rowBase + t + 8 * half;
    S2[(size_t)m * GCN_N_PAD + r] = acc[t];
  }
}

// ---------------------------------------------------------------------------
// out init: out[i*10+c] = b2[c]
// ---------------------------------------------------------------------------
__global__ __launch_bounds__(256) void gcn_init_out(
    float* __restrict__ out, const float* __restrict__ b2, int n)
{
  const int i = blockIdx.x * 256 + threadIdx.x;
  if (i < n) out[i] = b2[i % GCN_N_CLASSES];
}

__global__ __launch_bounds__(256) void gcn_zero(float* __restrict__ p, int n)
{
  const int i = blockIdx.x * 256 + threadIdx.x;
  if (i < n) p[i] = 0.0f;
}

// ---------------------------------------------------------------------------
// SpMM layer 2: out[dst*10+c] += w_e * S2[src*16+c]; 16 threads per edge.
// ---------------------------------------------------------------------------
__global__ __launch_bounds__(256) void gcn_spmm2(
    const int* __restrict__ src, const int* __restrict__ dst,
    const float* __restrict__ w, const float* __restrict__ S2,
    float* __restrict__ out, int nedges)
{
  const int idx = blockIdx.x * 256 + threadIdx.x;
  const int e = idx >> 4;
  const int c = idx & 15;
  if (e >= nedges || c >= GCN_N_CLASSES) return;
  const float v = w[e] * S2[(size_t)src[e] * GCN_N_PAD + c];
  atomicAdd(out + (size_t)dst[e] * GCN_N_CLASSES + c, v);
}

// ---------------------------------------------------------------------------
extern "C" void kernel_launch(void* const* d_in, const int* in_sizes, int n_in,
                              void* d_out, int out_size, void* d_ws, size_t ws_size,
                              hipStream_t stream) {
  const float* x  = (const float*)d_in[0];   // [N,512]
  const int*   es = (const int*)  d_in[1];   // [E]
  const int*   ed = (const int*)  d_in[2];   // [E]
  const float* ew = (const float*)d_in[3];   // [E]
  const float* W1 = (const float*)d_in[4];   // [512,128]
  const float* b1 = (const float*)d_in[5];   // [128]
  const float* W2 = (const float*)d_in[6];   // [128,10]
  const float* b2 = (const float*)d_in[7];   // [10]
  float* out = (float*)d_out;                // [N,10]

  (void)in_sizes; (void)n_in; (void)out_size; (void)ws_size;

  // Workspace carve (row-padded)
  float* S1  = (float*)d_ws;                                   // R1*128
  float* AGG = S1 + (size_t)GCN_ROWS1_PAD * GCN_N_HID;         // R2*128
  float* S2  = AGG + (size_t)GCN_ROWS2_PAD * GCN_N_HID;        // R2*16

  const int aggN = GCN_ROWS2_PAD * GCN_N_HID;
  const int outN = GCN_N_NODES * GCN_N_CLASSES;

  // Layer 1 GEMM (LDS-staged WMMA bf16): 64-row blocks, padded stores
  gcn_gemm1_wmma<<<GCN_ROWS1_PAD / G1_BM, 256, 0, stream>>>(
      x, W1, S1, GCN_N_NODES);

  // Zero (padded) aggregation buffer, then edge scatter-add
  gcn_zero<<<(aggN + 255) / 256, 256, 0, stream>>>(AGG, aggN);
  gcn_spmm1<<<(GCN_N_EDGES + 7) / 8, 256, 0, stream>>>(es, ed, ew, S1, AGG,
                                                       GCN_N_EDGES);

  // Layer 2 GEMM (LDS-staged WMMA bf16), relu(agg+b1) fused into A load
  gcn_gemm2_wmma<<<GCN_ROWS2_PAD / G2_BM, 256, 0, stream>>>(AGG, b1, W2, S2);

  // out = b2 + scatter-add
  gcn_init_out<<<(outN + 255) / 256, 256, 0, stream>>>(out, b2, outN);
  gcn_spmm2<<<((GCN_N_EDGES * 16) + 255) / 256, 256, 0, stream>>>(
      es, ed, ew, S2, out, GCN_N_EDGES);
}
